// MoEBlock_3770981286053
// MI455X (gfx1250) — compile-verified, matrix operand order checked by reading, not run on hardware
//
#include <hip/hip_runtime.h>
#include <hip/hip_bf16.h>

typedef __attribute__((ext_vector_type(16))) _Float16 v16h;
typedef __attribute__((ext_vector_type(8)))  _Float16 v8h;
typedef __attribute__((ext_vector_type(8)))  float    v8f;

#define NE 5
#define NK 3
#define NB 16
#define NC 128
#define NH 64
#define NW 64
#define NHW 4096
#define NDW 256
#define LEPS 1e-6f

// Build a 16x32 f16 WMMA operand fragment per CDNA5 ISA layout:
// lane = 16*h + r; VGPR slots 0-7 hold K = 8h..8h+7 (contiguous 16B),
// slots 8-15 hold K = 16+8h..16+8h+7 (contiguous 16B) -> two b128 loads.
__device__ __forceinline__ v16h load_frag(const _Float16* p) {
    v8h a = *(const v8h*)p;
    v8h b = *(const v8h*)(p + 16);
    return __builtin_shufflevector(a, b, 0,1,2,3,4,5,6,7,8,9,10,11,12,13,14,15);
}

// ---------------- gate / counts / weights passthrough ----------------
__global__ void k_gate(const float* __restrict__ w, float* __restrict__ gate,
                       float* __restrict__ tail) {
    __shared__ int cnt[NE];
    int t = threadIdx.x;
    if (t < NE) cnt[t] = 0;
    __syncthreads();
    if (t < NB) {
        float v[NE]; bool used[NE];
        for (int e = 0; e < NE; ++e) { v[e] = w[t*NE + e]; used[e] = false; gate[t*NE + e] = 0.f; }
        int ti[NK]; float tv[NK]; float s = 0.f;
        for (int k = 0; k < NK; ++k) {
            float best = -1e30f; int bi = 0;
            for (int e = 0; e < NE; ++e)
                if (!used[e] && v[e] > best) { best = v[e]; bi = e; }
            used[bi] = true; ti[k] = bi; tv[k] = best; s += best;
        }
        for (int k = 0; k < NK; ++k) {
            gate[t*NE + ti[k]] = tv[k] / s;
            atomicAdd(&cnt[ti[k]], 1);
        }
    }
    __syncthreads();
    if (t < NE) ((int*)tail)[t] = cnt[t];              // counts (int32 bit pattern)
    for (int i = t; i < NB*NE; i += blockDim.x) tail[NE + i] = w[i];  // weights
}

// ---------------- f32 -> f16 weight conversion (per expert) ----------------
__global__ void k_wcvt(const float* __restrict__ s1, const float* __restrict__ s3,
                       const float* __restrict__ s4, const float* __restrict__ s5,
                       _Float16* __restrict__ dst) {
    int i = blockIdx.x * 256 + threadIdx.x;   // 0..98303
    float v;
    if      (i < 32768) v = s1[i];            // conv1 256x128
    else if (i < 49152) v = s3[i - 32768];    // conv3 128x128
    else if (i < 81920) v = s4[i - 49152];    // conv4 256x128
    else                v = s5[i - 81920];    // conv5 128x128
    dst[i] = (_Float16)v;
}

// ---------------- LN1 + conv1 (M=256,K=128) GEMM, 128-pixel tiles ----------------
__global__ __launch_bounds__(256) void k_ln1gemm(
    const float* __restrict__ feat, const _Float16* __restrict__ wc1,
    const float* __restrict__ b1, const float* __restrict__ ln1w,
    const float* __restrict__ ln1b, float* __restrict__ y1) {
    __shared__ __align__(16) float    fbuf[128*130];
    __shared__ __align__(16) _Float16 bbuf[128*136];
    __shared__ float red[2*256];
    __shared__ float muS[128], rsS[128];
    int t = threadIdx.x;
    int b = blockIdx.x >> 5;
    int pixBase = (blockIdx.x & 31) * 128;
    {   // coalesced channel-major load of the pixel tile
        int p = t & 127, cq = t >> 7;
        for (int it = 0; it < 64; ++it) {
            int c = it*2 + cq;
            fbuf[p*130 + c] = feat[((size_t)b*NC + c)*NHW + pixBase + p];
        }
    }
    __syncthreads();
    {   // per-pixel LayerNorm stats (2 threads per pixel)
        int p = t & 127, q = t >> 7;
        float s = 0.f, ss = 0.f;
        for (int k = q*64; k < q*64 + 64; ++k) { float v = fbuf[p*130 + k]; s += v; ss += v*v; }
        red[q*128 + p] = s; red[256 + q*128 + p] = ss;
    }
    __syncthreads();
    if (t < 128) {
        float s = red[t] + red[128 + t];
        float ss = red[256 + t] + red[384 + t];
        float mu = s * (1.0f/NC), var = ss * (1.0f/NC) - mu*mu;
        muS[t] = mu; rsS[t] = rsqrtf(var + LEPS);
    }
    __syncthreads();
    {   // normalize -> f16 B operand [pixel][channel]
        int p = t & 127, q = t >> 7;
        float mu = muS[p], rs = rsS[p];
        for (int k = q*64; k < q*64 + 64; ++k)
            bbuf[p*136 + k] = (_Float16)((fbuf[p*130 + k] - mu)*rs*ln1w[k] + ln1b[k]);
    }
    __syncthreads();
    // GEMM: 16 m-tiles x 8 n-tiles, 8 waves; A fragments + bias hoisted per m-tile
    int wid = t >> 5, lane = t & 31, ln = lane & 15, h = lane >> 4;
    for (int tmi = 0; tmi < 2; ++tmi) {
        int tm = wid*2 + tmi;
        v16h Af[4];
        #pragma unroll
        for (int kq = 0; kq < 4; ++kq)
            Af[kq] = load_frag(wc1 + (size_t)(tm*16 + ln)*NC + kq*32 + 8*h);
        int m0 = tm*16 + 8*h;
        float bias[8];
        #pragma unroll
        for (int v = 0; v < 8; ++v) bias[v] = b1[m0 + v];
        for (int tn = 0; tn < 8; ++tn) {
            v8f acc = {};
            #pragma unroll
            for (int kq = 0; kq < 4; ++kq) {
                v16h Bf = load_frag(bbuf + (tn*16 + ln)*136 + kq*32 + 8*h);
                acc = __builtin_amdgcn_wmma_f32_16x16x32_f16(false, Af[kq], false, Bf,
                                                             (short)0, acc, false, false);
            }
            int n = pixBase + tn*16 + ln;
            #pragma unroll
            for (int v = 0; v < 8; ++v)
                y1[((size_t)b*NDW + (m0 + v))*NHW + n] = acc[v] + bias[v];
        }
    }
}

// ------- depthwise 3x3 + SimpleGate + deterministic partial sums -------
__global__ __launch_bounds__(256) void k_dwsg(
    const float* __restrict__ y1, const float* __restrict__ w2,
    const float* __restrict__ b2, float* __restrict__ tb, float* __restrict__ part) {
    __shared__ float s0[18*66], s1[18*66];
    __shared__ float sred[256];
    int t = threadIdx.x;
    int c  = blockIdx.x & 127;
    int rb = (blockIdx.x >> 7) & 3;
    int b  = blockIdx.x >> 9;
    const float* p0 = y1 + ((size_t)b*NDW + c)*NHW;
    const float* p1 = y1 + ((size_t)b*NDW + c + NC)*NHW;
    for (int idx = t; idx < 18*66; idx += 256) {
        int r = idx / 66, col = idx % 66;
        int gr = rb*16 + r - 1, gc = col - 1;
        bool ok = (gr >= 0) && (gr < NH) && (gc >= 0) && (gc < NW);
        s0[idx] = ok ? p0[gr*NW + gc] : 0.f;
        s1[idx] = ok ? p1[gr*NW + gc] : 0.f;
    }
    float wa[9], wg[9];
    #pragma unroll
    for (int j = 0; j < 9; ++j) { wa[j] = w2[c*9 + j]; wg[j] = w2[(c + NC)*9 + j]; }
    float ba = b2[c], bg = b2[c + NC];
    __syncthreads();
    float ls = 0.f;
    float* tout = tb + ((size_t)b*NC + c)*NHW;
    for (int i = 0; i < 4; ++i) {
        int idx = t + i*256;
        int r = idx >> 6, col = idx & 63;
        float a = ba, g = bg;
        #pragma unroll
        for (int dy = 0; dy < 3; ++dy)
            #pragma unroll
            for (int dx = 0; dx < 3; ++dx) {
                a += wa[dy*3 + dx] * s0[(r + dy)*66 + col + dx];
                g += wg[dy*3 + dx] * s1[(r + dy)*66 + col + dx];
            }
        float tv = a * g;
        tout[(rb*16 + r)*NW + col] = tv;
        ls += tv;
    }
    sred[t] = ls; __syncthreads();
    for (int off = 128; off > 0; off >>= 1) {
        if (t < off) sred[t] += sred[t + off];
        __syncthreads();
    }
    if (t == 0) part[((size_t)b*NC + c)*4 + rb] = sred[0];
}

// ---------------- SCA: mean + 128x128 matvec ----------------
__global__ void k_sca(const float* __restrict__ part, const float* __restrict__ scaw,
                      const float* __restrict__ scab, float* __restrict__ sbuf) {
    __shared__ float m[NC];
    int b = blockIdx.x, c = threadIdx.x;
    const float* pp = part + ((size_t)b*NC + c)*4;
    m[c] = (pp[0] + pp[1] + pp[2] + pp[3]) * (1.0f/NHW);
    __syncthreads();
    float acc = scab[c];
    const float* wr = scaw + c*NC;
    for (int k = 0; k < NC; ++k) acc += wr[k] * m[k];
    sbuf[b*NC + c] = acc;
}

// --- fused tail: *SCA -> conv3 -> +x*beta -> LN2 -> conv4 -> SG -> conv5 -> gated out ---
__global__ __launch_bounds__(256) void k_tail(
    const float* __restrict__ feat, const float* __restrict__ tb,
    const float* __restrict__ sbuf, const _Float16* __restrict__ wc3,
    const float* __restrict__ b3, const float* __restrict__ beta,
    const float* __restrict__ ln2w, const float* __restrict__ ln2b,
    const _Float16* __restrict__ wc4, const float* __restrict__ b4,
    const _Float16* __restrict__ wc5, const float* __restrict__ b5,
    const float* __restrict__ gamma, const float* __restrict__ gate,
    int e, int first, float* __restrict__ out) {
    __shared__ __align__(16) _Float16 bbuf[64*136];
    __shared__ __align__(16) float    x1[64*130];
    __shared__ __align__(16) _Float16 mid[64*264];
    __shared__ float red[2*256];
    __shared__ float muS[64], rsS[64];
    int t = threadIdx.x;
    int b = blockIdx.x >> 6;
    int pixBase = (blockIdx.x & 63) * 64;
    {   // load t*s (f16 B operand) and feat (residual base)
        int p = t & 63, cq = t >> 6;
        for (int it = 0; it < 32; ++it) {
            int c = it*4 + cq;
            float tv = tb[((size_t)b*NC + c)*NHW + pixBase + p] * sbuf[b*NC + c];
            bbuf[p*136 + c] = (_Float16)tv;
            x1[p*130 + c]   = feat[((size_t)b*NC + c)*NHW + pixBase + p];
        }
    }
    __syncthreads();
    int wid = t >> 5, lane = t & 31, ln = lane & 15, h = lane >> 4;
    // GEMM1: conv3 (M=128, tm = wid) ; x1 += (u + b3)*beta
    {
        int tm = wid;
        v16h Af[4];
        #pragma unroll
        for (int kq = 0; kq < 4; ++kq)
            Af[kq] = load_frag(wc3 + (size_t)(tm*16 + ln)*NC + kq*32 + 8*h);
        int m0 = tm*16 + 8*h;
        float bias[8], bet[8];
        #pragma unroll
        for (int v = 0; v < 8; ++v) { bias[v] = b3[m0 + v]; bet[v] = beta[m0 + v]; }
        for (int tn = 0; tn < 4; ++tn) {
            v8f acc = {};
            #pragma unroll
            for (int kq = 0; kq < 4; ++kq) {
                v16h Bf = load_frag(bbuf + (tn*16 + ln)*136 + kq*32 + 8*h);
                acc = __builtin_amdgcn_wmma_f32_16x16x32_f16(false, Af[kq], false, Bf,
                                                             (short)0, acc, false, false);
            }
            int n = tn*16 + ln;
            #pragma unroll
            for (int v = 0; v < 8; ++v)
                x1[n*130 + (m0 + v)] += (acc[v] + bias[v]) * bet[v];
        }
    }
    __syncthreads();
    {   // LN2 stats (4 threads per pixel)
        int p = t & 63, q = t >> 6;
        float s = 0.f, ss = 0.f;
        for (int k = q*32; k < q*32 + 32; ++k) { float v = x1[p*130 + k]; s += v; ss += v*v; }
        red[q*64 + p] = s; red[256 + q*64 + p] = ss;
    }
    __syncthreads();
    if (t < 64) {
        float s  = red[t] + red[64 + t] + red[128 + t] + red[192 + t];
        float ss = red[256 + t] + red[320 + t] + red[384 + t] + red[448 + t];
        float mu = s * (1.0f/NC), var = ss * (1.0f/NC) - mu*mu;
        muS[t] = mu; rsS[t] = rsqrtf(var + LEPS);
    }
    __syncthreads();
    {
        int p = t & 63, q = t >> 6;
        float mu = muS[p], rs = rsS[p];
        for (int k = q*32; k < q*32 + 32; ++k)
            bbuf[p*136 + k] = (_Float16)((x1[p*130 + k] - mu)*rs*ln2w[k] + ln2b[k]);
    }
    __syncthreads();
    // GEMM2: conv4 (M=256, tm = 2*wid+tmi) -> mid (f16)
    for (int tmi = 0; tmi < 2; ++tmi) {
        int tm = wid*2 + tmi;
        v16h Af[4];
        #pragma unroll
        for (int kq = 0; kq < 4; ++kq)
            Af[kq] = load_frag(wc4 + (size_t)(tm*16 + ln)*NC + kq*32 + 8*h);
        int m0 = tm*16 + 8*h;
        float bias[8];
        #pragma unroll
        for (int v = 0; v < 8; ++v) bias[v] = b4[m0 + v];
        for (int tn = 0; tn < 4; ++tn) {
            v8f acc = {};
            #pragma unroll
            for (int kq = 0; kq < 4; ++kq) {
                v16h Bf = load_frag(bbuf + (tn*16 + ln)*136 + kq*32 + 8*h);
                acc = __builtin_amdgcn_wmma_f32_16x16x32_f16(false, Af[kq], false, Bf,
                                                             (short)0, acc, false, false);
            }
            int n = tn*16 + ln;
            #pragma unroll
            for (int v = 0; v < 8; ++v)
                mid[n*264 + (m0 + v)] = (_Float16)(acc[v] + bias[v]);
        }
    }
    __syncthreads();
    // SimpleGate: bbuf[n][k] = mid[n][k] * mid[n][k+128]
    for (int i = 0; i < 32; ++i) {
        int idx = t + i*256;
        int n = idx >> 7, k = idx & 127;
        bbuf[n*136 + k] = (_Float16)((float)mid[n*264 + k] * (float)mid[n*264 + NC + k]);
    }
    __syncthreads();
    // GEMM3: conv5 (M=128, tm = wid) ; out (+)= gate * (x1 + (z + b5)*gamma)
    float g = gate[b*NE + e];
    {
        int tm = wid;
        v16h Af[4];
        #pragma unroll
        for (int kq = 0; kq < 4; ++kq)
            Af[kq] = load_frag(wc5 + (size_t)(tm*16 + ln)*NC + kq*32 + 8*h);
        int m0 = tm*16 + 8*h;
        float bias[8], gam[8];
        #pragma unroll
        for (int v = 0; v < 8; ++v) { bias[v] = b5[m0 + v]; gam[v] = gamma[m0 + v]; }
        for (int tn = 0; tn < 4; ++tn) {
            v8f acc = {};
            #pragma unroll
            for (int kq = 0; kq < 4; ++kq) {
                v16h Bf = load_frag(bbuf + (tn*16 + ln)*136 + kq*32 + 8*h);
                acc = __builtin_amdgcn_wmma_f32_16x16x32_f16(false, Af[kq], false, Bf,
                                                             (short)0, acc, false, false);
            }
            int n = tn*16 + ln;
            #pragma unroll
            for (int v = 0; v < 8; ++v) {
                int mm = m0 + v;
                float val = x1[n*130 + mm] + (acc[v] + bias[v]) * gam[v];
                size_t o = ((size_t)b*NC + mm)*NHW + pixBase + n;
                if (first) out[o] = g * val; else out[o] += g * val;
            }
        }
    }
}

extern "C" void kernel_launch(void* const* d_in, const int* in_sizes, int n_in,
                              void* d_out, int out_size, void* d_ws, size_t ws_size,
                              hipStream_t stream) {
    const float* feat    = (const float*)d_in[0];
    const float* weights = (const float*)d_in[1];
    const float* ln1w = (const float*)d_in[2];  const float* ln1b = (const float*)d_in[3];
    const float* c1w  = (const float*)d_in[4];  const float* c1b  = (const float*)d_in[5];
    const float* c2w  = (const float*)d_in[6];  const float* c2b  = (const float*)d_in[7];
    const float* scaw = (const float*)d_in[8];  const float* scab = (const float*)d_in[9];
    const float* c3w  = (const float*)d_in[10]; const float* c3b  = (const float*)d_in[11];
    const float* ln2w = (const float*)d_in[12]; const float* ln2b = (const float*)d_in[13];
    const float* c4w  = (const float*)d_in[14]; const float* c4b  = (const float*)d_in[15];
    const float* c5w  = (const float*)d_in[16]; const float* c5b  = (const float*)d_in[17];
    const float* beta = (const float*)d_in[18]; const float* gamma = (const float*)d_in[19];
    float* out  = (float*)d_out;
    float* tail = out + (size_t)NB*NC*NHW;   // counts[5] (int bits) then weights[80]

    // workspace layout (~96 MiB; y1/t stay resident in the 192MB L2 between kernels)
    char* ws = (char*)d_ws;
    _Float16* w16   = (_Float16*)ws;            // 98304 halves
    _Float16* w16c1 = w16;
    _Float16* w16c3 = w16 + 32768;
    _Float16* w16c4 = w16 + 49152;
    _Float16* w16c5 = w16 + 81920;
    float* gateB = (float*)(ws + 196608);       // 80 f32
    float* sbuf  = (float*)(ws + 197120);       // 16*128 f32
    float* part  = (float*)(ws + 205312);       // 16*128*4 f32
    float* tb    = (float*)(ws + 238080);       // 16*128*4096 f32 (33.5MB)
    float* y1    = (float*)(ws + 33792512);     // 16*256*4096 f32 (67MB)

    k_gate<<<1, 128, 0, stream>>>(weights, gateB, tail);
    for (int e = 0; e < NE; ++e) {
        k_wcvt<<<384, 256, 0, stream>>>(c1w + (size_t)e*NDW*NC, c3w + (size_t)e*NC*NC,
                                        c4w + (size_t)e*NDW*NC, c5w + (size_t)e*NC*NC, w16);
        k_ln1gemm<<<NB*32, 256, 0, stream>>>(feat, w16c1, c1b + e*NDW,
                                             ln1w + e*NC, ln1b + e*NC, y1);
        k_dwsg<<<NB*NC*4, 256, 0, stream>>>(y1, c2w + (size_t)e*NDW*9, c2b + e*NDW, tb, part);
        k_sca<<<NB, NC, 0, stream>>>(part, scaw + (size_t)e*NC*NC, scab + e*NC, sbuf);
        k_tail<<<NB*64, 256, 0, stream>>>(feat, tb, sbuf, w16c3, c3b + e*NC, beta + e*NC,
                                          ln2w + e*NC, ln2b + e*NC, w16c4, c4b + e*NDW,
                                          w16c5, c5b + e*NC, gamma + e*NC,
                                          gateB, e, (e == 0) ? 1 : 0, out);
    }
}
